// AttentionalCopula_2267742732410
// MI455X (gfx1250) — matrix operand order repaired and checked by reading, not compiled, not run on hardware
//
#include <hip/hip_runtime.h>
#include <math.h>

// ---------------------------------------------------------------------------
// Problem constants: B=8, S=16, T=64, F=256, V=1024, P=512,
// NHEAD=8, ADIM=64, E=512, MLPD=512, NLAYERS=2, RES=100.
// ---------------------------------------------------------------------------
#define BATCH  8
#define VTOK   1024
#define FDIM   256
#define PPTS   512
#define NHEAD  8
#define ADIM   64
#define EDIM   512
#define MLPD   512
#define NLAY   2
#define RES    100
#define NNBR   32      // 2*S

typedef __attribute__((ext_vector_type(16))) _Float16 v16h;
typedef __attribute__((ext_vector_type(8)))  _Float16 v8h;
typedef __attribute__((ext_vector_type(8)))  float    v8f;
typedef __attribute__((ext_vector_type(4)))  float    v4f;
typedef __attribute__((ext_vector_type(4)))  int      v4i;

// ---- gfx1250 async global->LDS copy (ASYNCcnt path), guarded ----
#ifndef __has_builtin
#define __has_builtin(x) 0
#endif
#if __has_builtin(__builtin_amdgcn_global_load_async_to_lds_b128)
#define USE_ASYNC_LDS 1
#else
#define USE_ASYNC_LDS 0
#endif

#define AS1 __attribute__((address_space(1)))
#define AS3 __attribute__((address_space(3)))

__device__ __forceinline__ void async_wait0() {
#if __has_builtin(__builtin_amdgcn_s_wait_asynccnt)
  __builtin_amdgcn_s_wait_asynccnt(0);
#else
  asm volatile("s_wait_asynccnt 0x0" ::: "memory");
#endif
}

// ---------------------------------------------------------------------------
// Tiled WMMA GEMM:  C[M,N] = act( A[M,K] * W[K,N] + bias[N] )
// Requirements: M % 128 == 0 (true for all call sites: 8192 / 4096).
// fp32 in/out, f16 multiply via v_wmma_f32_16x16x32_f16, fp32 accumulate.
// Block = 256 threads = 8 waves; tile 128(M) x 64(N); K stepped by 32.
// A tile: global_load_async_to_lds_b128 (DMA overlapped with B staging),
// converted to f16 at fragment build. B tile staged f16 N-major in LDS.
// ---------------------------------------------------------------------------
#define BM 128
#define BN 64
#define BK 32
#define APITCH 36   // floats: 32 + 4 pad (144B rows; 16B aligned, bank-skewed)
#define BPITCH 40   // halfs : 32 + 8 pad (80B rows; 16B aligned)

__global__ __launch_bounds__(256)
void gemm_wmma_f16(const float* __restrict__ A, const float* __restrict__ W,
                   const float* __restrict__ bias, float* __restrict__ C,
                   int M, int N, int K, int ldc, int relu) {
  __shared__ __align__(16) float    As[BM][APITCH];   // row-major M x K (fp32)
  __shared__ __align__(16) _Float16 Bs[BN][BPITCH];   // N-major: Bs[n][k] (f16)

  const int tid   = threadIdx.x;
  const int lane  = tid & 31;
  const int wave  = tid >> 5;          // 0..7
  const int waveM = wave >> 1;         // 0..3 -> rows waveM*32
  const int waveN = wave & 1;          // 0..1 -> cols waveN*32

  const int rowBase = blockIdx.y * BM;
  const int colBase = blockIdx.x * BN;

  v8f acc[2][2] = {};

  const int m16 = lane & 15;
  const int kg  = lane >> 4;           // A fragment k-group
  const int n16 = lane & 15;
  const int kh  = lane >> 4;           // B fragment k-half

  for (int k0 = 0; k0 < K; k0 += BK) {
    const bool ktail = (k0 + BK > K);  // uniform across block

    // ---- stage A tile (128 x 32 fp32) ----
    if (!ktail) {
#if USE_ASYNC_LDS
      // CDNA5 async DMA: each lane moves one 16B chunk straight into LDS.
      // Wait is deferred until after B staging so DMA overlaps VMEM work.
      #pragma unroll
      for (int i = 0; i < (BM * BK) / (256 * 4); ++i) {   // 4 chunks/thread
        int idx = tid + i * 256;
        int r = idx >> 3, c4 = idx & 7;                   // 8 float4 per row
        const float* gp = A + (size_t)(rowBase + r) * K + k0 + c4 * 4;
        __builtin_amdgcn_global_load_async_to_lds_b128(
            (AS1 v4i*)gp, (AS3 v4i*)&As[r][c4 * 4], 0, 0);
      }
#else
      #pragma unroll
      for (int i = 0; i < (BM * BK) / (256 * 4); ++i) {
        int idx = tid + i * 256;
        int r = idx >> 3, c4 = idx & 7;
        const float* gp = A + (size_t)(rowBase + r) * K + k0 + c4 * 4;
        As[r][c4 * 4 + 0] = gp[0];
        As[r][c4 * 4 + 1] = gp[1];
        As[r][c4 * 4 + 2] = gp[2];
        As[r][c4 * 4 + 3] = gp[3];
      }
#endif
    } else {
      // K tail: two phases (all loads, then all stores) so loads pipeline
      float raw[(BM * BK) / 256];
      #pragma unroll
      for (int i = 0; i < (BM * BK) / 256; ++i) {
        int idx = tid + i * 256;
        int r = idx >> 5, c = idx & 31;
        int gk = k0 + c;
        raw[i] = A[(size_t)(rowBase + r) * K + (gk < K ? gk : K - 1)];
      }
      #pragma unroll
      for (int i = 0; i < (BM * BK) / 256; ++i) {
        int idx = tid + i * 256;
        int r = idx >> 5, c = idx & 31;
        int gk = k0 + c;
        As[r][c] = (gk < K) ? raw[i] : 0.0f;
      }
    }

    // ---- stage W tile (32 x 64 fp32 -> f16), transposed into N-major LDS ----
    #pragma unroll
    for (int i = 0; i < (BK * BN) / 256; ++i) {
      int idx = tid + i * 256;
      int k = idx >> 6, n = idx & 63;
      int gk = k0 + k, gn = colBase + n;
      int gkc = gk < K ? gk : K - 1;
      int gnc = gn < N ? gn : N - 1;
      float v = W[(size_t)gkc * N + gnc];
      v = (gk < K && gn < N) ? v : 0.0f;
      Bs[n][k] = (_Float16)v;
    }

#if USE_ASYNC_LDS
    if (!ktail) async_wait0();           // A-tile DMA done (overlapped above)
#endif
    __syncthreads();

    // ---- build fragments and issue WMMA ----
    v16h afrag[2], bfrag[2];
    #pragma unroll
    for (int mi = 0; mi < 2; ++mi) {
      int ar = waveM * 32 + mi * 16 + m16;
      v4f a0 = *reinterpret_cast<const v4f*>(&As[ar][kg * 8]);        // K=kg*8..
      v4f a1 = *reinterpret_cast<const v4f*>(&As[ar][kg * 8 + 4]);
      v4f a2 = *reinterpret_cast<const v4f*>(&As[ar][16 + kg * 8]);   // K=16+kg*8..
      v4f a3 = *reinterpret_cast<const v4f*>(&As[ar][16 + kg * 8 + 4]);
      v16h a;
      #pragma unroll
      for (int j = 0; j < 4; ++j) {
        a[j]      = (_Float16)a0[j];
        a[4 + j]  = (_Float16)a1[j];
        a[8 + j]  = (_Float16)a2[j];
        a[12 + j] = (_Float16)a3[j];
      }
      afrag[mi] = a;
    }
    #pragma unroll
    for (int ni = 0; ni < 2; ++ni) {
      int br = waveN * 32 + ni * 16 + n16;
      v8h lo = *reinterpret_cast<const v8h*>(&Bs[br][kh * 8]);       // K = kh*8..+7
      v8h hi = *reinterpret_cast<const v8h*>(&Bs[br][16 + kh * 8]);  // K = 16+kh*8..+7
      bfrag[ni] = __builtin_shufflevector(lo, hi, 0,1,2,3,4,5,6,7,8,9,10,11,12,13,14,15);
    }
    #pragma unroll
    for (int mi = 0; mi < 2; ++mi)
      #pragma unroll
      for (int ni = 0; ni < 2; ++ni)
        acc[mi][ni] = __builtin_amdgcn_wmma_f32_16x16x32_f16(
            false, afrag[mi], false, bfrag[ni], (short)0, acc[mi][ni], false, false);
    __syncthreads();
  }

  // ---- epilogue: bias + optional ReLU (rows always in-bounds; N may tail) ----
  const int rh = lane >> 4;
  #pragma unroll
  for (int ni = 0; ni < 2; ++ni) {
    int col = colBase + waveN * 32 + ni * 16 + n16;
    float bv = bias[col < N ? col : 0];
    #pragma unroll
    for (int mi = 0; mi < 2; ++mi) {
      #pragma unroll
      for (int r = 0; r < 8; ++r) {
        int row = rowBase + waveM * 32 + mi * 16 + rh * 8 + r;
        float v = acc[mi][ni][r] + bv;
        if (relu) v = fmaxf(v, 0.0f);
        if (col < N) C[(size_t)row * ldc + col] = v;
      }
    }
  }
}

// ---------------------------------------------------------------------------
// merged[b*V+v] = concat(enc_row[0..255], u)   (257 wide)
// ---------------------------------------------------------------------------
__global__ void build_merged_kernel(const float* __restrict__ enc,
                                    const float* __restrict__ u,
                                    float* __restrict__ merged) {
  int row = blockIdx.x;          // b*V+v  (8192 rows)
  int tid = threadIdx.x;         // 256
  merged[(size_t)row * 257 + tid] = enc[(size_t)row * 256 + tid];
  if (tid == 0) merged[(size_t)row * 257 + 256] = u[row];
}

// avG[b*P+p][f] = enc[b*V + pred[p]][f]
__global__ void gather_pred_kernel(const float* __restrict__ enc,
                                   const int* __restrict__ pred,
                                   float* __restrict__ out) {
  int row = blockIdx.x;          // b*P+p (4096 rows)
  int b = row >> 9, p = row & 511;
  int tid = threadIdx.x;         // 256
  out[(size_t)row * 256 + tid] =
      enc[((size_t)b * VTOK + pred[p]) * 256 + tid];
}

// ---------------------------------------------------------------------------
// Attention: one wave32 block per (b,p,h). Lane = neighbor n (0..31).
// keys/vals layout: [(h*B + b)*V + idx][ADIM]
// ---------------------------------------------------------------------------
__global__ __launch_bounds__(32)
void attention_kernel(const float* __restrict__ av,
                      const float* __restrict__ keys,
                      const float* __restrict__ vals,
                      const int* __restrict__ nbr,
                      const int* __restrict__ pmask,
                      float* __restrict__ att) {
  int gid = blockIdx.x;                 // ((b*P + p)*H + h)
  int h = gid & 7;
  int p = (gid >> 3) & 511;
  int b = gid >> 12;
  int lane = threadIdx.x;

  int   idx = nbr[p * NNBR + lane];
  float m   = (float)pmask[p * NNBR + lane];

  const float* q  = av   + ((size_t)(b * PPTS + p)) * EDIM + h * ADIM;
  const float* kp = keys + (((size_t)(h * BATCH + b)) * VTOK + idx) * ADIM;

  float s = 0.0f;
  #pragma unroll 8
  for (int d = 0; d < ADIM; ++d) s += q[d] * kp[d];
  s *= m * 0.125f;                       // mask then 1/sqrt(64)

  // wave32 softmax over 32 neighbors (masked entries keep score 0, like ref)
  float mx = s;
  #pragma unroll
  for (int o = 16; o > 0; o >>= 1) mx = fmaxf(mx, __shfl_xor(mx, o, 32));
  float e = __expf(s - mx);
  float sum = e;
  #pragma unroll
  for (int o = 16; o > 0; o >>= 1) sum += __shfl_xor(sum, o, 32);
  float w = e / sum;

  __shared__ float ws[NNBR];
  __shared__ int   is[NNBR];
  ws[lane] = w * m;                      // fold value-mask into weight
  is[lane] = idx;
  __syncthreads();

  for (int d = lane; d < ADIM; d += 32) {
    float a = 0.0f;
    #pragma unroll 8
    for (int n = 0; n < NNBR; ++n)
      a += ws[n] * vals[(((size_t)(h * BATCH + b)) * VTOK + is[n]) * ADIM + d];
    att[((size_t)(b * PPTS + p)) * EDIM + h * ADIM + d] = a;
  }
}

// ---------------------------------------------------------------------------
// out = LayerNorm(x + r) * g + b   over E=512; one 256-thread block per row.
// Safe for out == x (each thread reads then writes only its own elements).
// ---------------------------------------------------------------------------
__global__ __launch_bounds__(256)
void add_ln_kernel(const float* __restrict__ x, const float* __restrict__ r,
                   const float* __restrict__ g, const float* __restrict__ bb,
                   float* __restrict__ out) {
  const int E = EDIM;
  int row = blockIdx.x, tid = threadIdx.x;
  __shared__ float red[256];

  float t0 = x[(size_t)row * E + tid]       + r[(size_t)row * E + tid];
  float t1 = x[(size_t)row * E + tid + 256] + r[(size_t)row * E + tid + 256];

  red[tid] = t0 + t1;
  __syncthreads();
  for (int o = 128; o > 0; o >>= 1) { if (tid < o) red[tid] += red[tid + o]; __syncthreads(); }
  float mu = red[0] * (1.0f / E);
  __syncthreads();

  float d0 = t0 - mu, d1 = t1 - mu;
  red[tid] = d0 * d0 + d1 * d1;
  __syncthreads();
  for (int o = 128; o > 0; o >>= 1) { if (tid < o) red[tid] += red[tid + o]; __syncthreads(); }
  float inv = rsqrtf(red[0] * (1.0f / E) + 1e-5f);

  out[(size_t)row * E + tid]       = d0 * inv * g[tid]       + bb[tid];
  out[(size_t)row * E + tid + 256] = d1 * inv * g[tid + 256] + bb[tid + 256];
}

// ---------------------------------------------------------------------------
// Final: out[b*P+p] = -( log(RES) + logits[t] - logsumexp(logits) )
// with t = clip(floor(u[b, pred[p]] * RES), 0, RES-1)
// ---------------------------------------------------------------------------
__global__ __launch_bounds__(128)
void final_loss_kernel(const float* __restrict__ logits,
                       const float* __restrict__ u,
                       const int* __restrict__ pred,
                       float* __restrict__ out) {
  int row = blockIdx.x;                 // b*P+p
  int b = row >> 9, p = row & 511;
  int tid = threadIdx.x;                // 128
  const float* lg = logits + (size_t)row * RES;
  __shared__ float red[128];

  float mx = -3.402823e38f;
  for (int i = tid; i < RES; i += 128) mx = fmaxf(mx, lg[i]);
  red[tid] = mx;
  __syncthreads();
  for (int o = 64; o > 0; o >>= 1) { if (tid < o) red[tid] = fmaxf(red[tid], red[tid + o]); __syncthreads(); }
  mx = red[0];
  __syncthreads();

  float s = 0.0f;
  for (int i = tid; i < RES; i += 128) s += __expf(lg[i] - mx);
  red[tid] = s;
  __syncthreads();
  for (int o = 64; o > 0; o >>= 1) { if (tid < o) red[tid] += red[tid + o]; __syncthreads(); }

  if (tid == 0) {
    float lse = mx + __logf(red[0]);
    float uu = u[(size_t)b * VTOK + pred[p]];
    int t = (int)floorf(uu * (float)RES);
    t = t < 0 ? 0 : (t > RES - 1 ? RES - 1 : t);
    out[row] = -(__logf((float)RES) + lg[t] - lse);
  }
}

// ---------------------------------------------------------------------------
static inline void launch_gemm(const float* A, const float* W, const float* bias,
                               float* C, int M, int N, int K, int ldc, int relu,
                               hipStream_t s) {
  dim3 grid((N + BN - 1) / BN, (M + BM - 1) / BM, 1);
  gemm_wmma_f16<<<grid, dim3(256), 0, s>>>(A, W, bias, C, M, N, K, ldc, relu);
}

extern "C" void kernel_launch(void* const* d_in, const int* in_sizes, int n_in,
                              void* d_out, int out_size, void* d_ws, size_t ws_size,
                              hipStream_t stream) {
  (void)in_sizes; (void)n_in; (void)out_size; (void)ws_size;

  const float* encoded  = (const float*)d_in[0];   // (B,V,F) after reshape
  const float* true_u   = (const float*)d_in[1];   // (B,V)
  const int*   pred     = (const int*)  d_in[2];   // (P,)
  const int*   nbr      = (const int*)  d_in[3];   // (P,32)
  const int*   pmask    = (const int*)  d_in[4];   // (P,32)
  const float* ds_W     = (const float*)d_in[5];
  const float* ds_b     = (const float*)d_in[6];
  const float* kvW1[2]  = {(const float*)d_in[7],  (const float*)d_in[13]};
  const float* kvB1[2]  = {(const float*)d_in[8],  (const float*)d_in[14]};
  const float* kvW2[2]  = {(const float*)d_in[9],  (const float*)d_in[15]};
  const float* kvB2[2]  = {(const float*)d_in[10], (const float*)d_in[16]};
  const float* kvW3[2]  = {(const float*)d_in[11], (const float*)d_in[17]};
  const float* kvB3[2]  = {(const float*)d_in[12], (const float*)d_in[18]};
  const float* aln_g    = (const float*)d_in[19];
  const float* aln_b    = (const float*)d_in[20];
  const float* ffW1     = (const float*)d_in[21];
  const float* ffb1     = (const float*)d_in[22];
  const float* ffW2     = (const float*)d_in[23];
  const float* ffb2     = (const float*)d_in[24];
  const float* fln_g    = (const float*)d_in[25];
  const float* fln_b    = (const float*)d_in[26];
  const float* deW1     = (const float*)d_in[27];
  const float* deb1     = (const float*)d_in[28];
  const float* deW2     = (const float*)d_in[29];
  const float* deb2     = (const float*)d_in[30];
  const float* deW3     = (const float*)d_in[31];
  const float* deb3     = (const float*)d_in[32];

  const int BV = BATCH * VTOK;   // 8192
  const int BP = BATCH * PPTS;   // 4096

  // -------- workspace carve-up (fp32 elements) --------
  float* ws     = (float*)d_ws;
  float* merged = ws;                                   // 8192*257
  float* avG    = merged + (size_t)BV * 257;            // 4096*256
  float* av     = avG    + (size_t)BP * 256;            // 4096*512
  float* h1     = av     + (size_t)BP * EDIM;           // 8192*512
  float* h2     = h1     + (size_t)BV * MLPD;           // 8192*512
  float* keys   = h2     + (size_t)BV * MLPD;           // 8*8192*64
  float* vals   = keys   + (size_t)NHEAD * BV * ADIM;   // 8*8192*64
  float* att    = vals   + (size_t)NHEAD * BV * ADIM;   // 4096*512
  float* tmp1   = att    + (size_t)BP * EDIM;           // 4096*512
  float* tmp2   = tmp1   + (size_t)BP * EDIM;           // 4096*512
  float* logits = tmp2   + (size_t)BP * EDIM;           // 4096*100

  // -------- preprocessing --------
  build_merged_kernel<<<BV, 256, 0, stream>>>(encoded, true_u, merged);
  gather_pred_kernel<<<BP, 256, 0, stream>>>(encoded, pred, avG);
  launch_gemm(avG, ds_W, ds_b, av, BP, EDIM, FDIM, EDIM, 0, stream);

  // -------- layers --------
  for (int l = 0; l < NLAY; ++l) {
    for (int kv = 0; kv < 2; ++kv) {
      float* outKV = (kv == 0) ? keys : vals;
      for (int h = 0; h < NHEAD; ++h) {
        size_t lh = (size_t)l * NHEAD + h;
        launch_gemm(merged, kvW1[kv] + lh * 257 * MLPD, kvB1[kv] + lh * MLPD,
                    h1, BV, MLPD, 257, MLPD, 1, stream);
        launch_gemm(h1, kvW2[kv] + lh * MLPD * MLPD, kvB2[kv] + lh * MLPD,
                    h2, BV, MLPD, MLPD, MLPD, 1, stream);
        launch_gemm(h2, kvW3[kv] + lh * MLPD * ADIM, kvB3[kv] + lh * ADIM,
                    outKV + (size_t)h * BV * ADIM, BV, ADIM, MLPD, ADIM, 0, stream);
      }
    }
    attention_kernel<<<BP * NHEAD, 32, 0, stream>>>(av, keys, vals, nbr, pmask, att);
    add_ln_kernel<<<BP, 256, 0, stream>>>(av, att, aln_g + l * EDIM, aln_b + l * EDIM, av);

    launch_gemm(av,   ffW1 + (size_t)l * EDIM * EDIM, ffb1 + l * EDIM, tmp1, BP, EDIM, EDIM, EDIM, 1, stream);
    launch_gemm(tmp1, ffW2 + (size_t)l * EDIM * EDIM, ffb2 + l * EDIM, tmp2, BP, EDIM, EDIM, EDIM, 0, stream);
    add_ln_kernel<<<BP, 256, 0, stream>>>(av, tmp2, fln_g + l * EDIM, fln_b + l * EDIM, av);
  }

  // -------- decoder --------
  launch_gemm(av,   deW1, deb1, tmp1, BP, MLPD, EDIM, MLPD, 1, stream);
  launch_gemm(tmp1, deW2, deb2, tmp2, BP, MLPD, MLPD, MLPD, 1, stream);
  launch_gemm(tmp2, deW3, deb3, logits, BP, RES, MLPD, RES, 0, stream);

  final_loss_kernel<<<BP, 128, 0, stream>>>(logits, true_u, pred, (float*)d_out);
}
